// encoder_bead_4956392259719
// MI455X (gfx1250) — matrix operand order repaired and checked by reading, not compiled, not run on hardware
//
#include <hip/hip_runtime.h>

#define NNODES 20000
#define DEG    16
#define NSLICE 4
#define FDIM   64

typedef _Float16 v16h __attribute__((ext_vector_type(16)));
typedef _Float16 v8h  __attribute__((ext_vector_type(8)));
typedef float    v8f  __attribute__((ext_vector_type(8)));

__device__ __forceinline__ v8f wmma16(v16h a, v16h b, v8f c) {
    // D = A(16x32 f16) * B(32x16 f16) + C(16x16 f32)
    return __builtin_amdgcn_wmma_f32_16x16x32_f16(false, a, false, b, (short)0, c, false, false);
}

__device__ __forceinline__ v16h cat8(v8h lo, v8h hi) {
    return __builtin_shufflevector(lo, hi, 0,1,2,3,4,5,6,7,8,9,10,11,12,13,14,15);
}

// Load f32 [0..7] and [16..23] from global, convert to an f16 B-operand fragment.
__device__ __forceinline__ v16h load_row_f16(const float* __restrict__ p) {
    v16h r;
#pragma unroll
    for (int j = 0; j < 4; ++j) {
        float4 q = ((const float4*)p)[(j & 1) + (j >> 1) * 4];
        r[4*j+0] = (_Float16)q.x; r[4*j+1] = (_Float16)q.y;
        r[4*j+2] = (_Float16)q.z; r[4*j+3] = (_Float16)q.w;
    }
    return r;
}

// TRANS-unit activations: one v_exp_f32 + one v_rcp_f32 each (co-execute with WMMA).
__device__ __forceinline__ float sigm_(float x) {
    return __builtin_amdgcn_rcpf(1.0f + __expf(-x));
}
__device__ __forceinline__ float tanh_(float x) {
    return 1.0f - 2.0f * __builtin_amdgcn_rcpf(1.0f + __expf(2.0f * x));
}

__global__ __launch_bounds__(256)
void sage_lstm_layer(const float* __restrict__ hin,     // [N, S, F]
                     const int*   __restrict__ nbr,     // [N, D]
                     const float* __restrict__ ew,      // [N, D]
                     const float* __restrict__ Wih,     // [4F, F]
                     const float* __restrict__ Whh,     // [4F, F]
                     const float* __restrict__ bih,     // [4F]
                     const float* __restrict__ bhh,     // [4F]
                     const float* __restrict__ Wself,   // [F, F]
                     const float* __restrict__ Wneigh,  // [F, F]
                     const float* __restrict__ bout,    // [F]
                     float*       __restrict__ hout)    // [N, S, F]
{
    // double-buffered message / hidden tiles for 32 nodes (f16 A-operands)
    __shared__ __align__(16) _Float16 smA[2][32 * FDIM];   // 2 x 4 KB
    __shared__ __align__(16) _Float16 shA[2][32 * FDIM];   // 2 x 4 KB

    const int tid    = threadIdx.x;
    const int lane   = tid & 31;
    const int wave   = tid >> 5;
    const int mgroup = wave >> 2;      // which 16-node M tile (0/1)
    const int cgroup = wave & 3;       // which 16-feature column group (0..3)
    const int bid    = blockIdx.x;
    const int s      = bid & (NSLICE - 1);
    const int node0  = (bid >> 2) * 32;
    const size_t rowstride = (size_t)NSLICE * FDIM;

    const int lrow  = lane & 15;       // A-row / B-col / D-col within tile
    const int hi16  = lane >> 4;       // half-wave select
    const int koffB = hi16 * 8;        // B fragment K offset
    const int koffA = hi16 * 8;        // A fragment K offset (also D row offset)

    const int gnode = tid >> 3;        // gather node 0..31
    const int gfeat = (tid & 7) * 8;   // gather feature base (8 floats/thread)

    // ---- persistent B operands: 4 gate groups x 2 K chunks, Wih & Whh ----
    v16h Bih[4][2], Bhh[4][2];
    const int fcol = cgroup * 16 + lrow;           // feature column 0..63
#pragma unroll
    for (int g = 0; g < 4; ++g) {
        const int gr = g * 64 + fcol;              // gate row 0..255
#pragma unroll
        for (int kc = 0; kc < 2; ++kc) {
            Bih[g][kc] = load_row_f16(Wih + gr * FDIM + kc * 32 + koffB);
            Bhh[g][kc] = load_row_f16(Whh + gr * FDIM + kc * 32 + koffB);
        }
    }
    // per-lane fused biases (loop-invariant)
    const float sbi = bih[      fcol] + bhh[      fcol];
    const float sbf = bih[ 64 + fcol] + bhh[ 64 + fcol];
    const float sbg = bih[128 + fcol] + bhh[128 + fcol];
    const float sbo = bih[192 + fcol] + bhh[192 + fcol];

    // ---- prologue: zero h, gather m_0 into buffer 0 ----
    {
        v8h z = {};
        *(v8h*)(shA[0] + gnode * FDIM + gfeat) = z;
        const int e = (node0 + gnode) * DEG;
        const int nb = nbr[e];
        const float w = ew[e];
        const float4* src = (const float4*)(hin + (size_t)nb * rowstride + (size_t)s * FDIM + gfeat);
        float4 q0 = src[0], q1 = src[1];
        v8h hv;
        hv[0] = (_Float16)(q0.x * w); hv[1] = (_Float16)(q0.y * w);
        hv[2] = (_Float16)(q0.z * w); hv[3] = (_Float16)(q0.w * w);
        hv[4] = (_Float16)(q1.x * w); hv[5] = (_Float16)(q1.y * w);
        hv[6] = (_Float16)(q1.z * w); hv[7] = (_Float16)(q1.w * w);
        *(v8h*)(smA[0] + gnode * FDIM + gfeat) = hv;
    }
    __syncthreads();

    v8f cst = {};                                   // cell-state fragment (in registers)
    int cur = 0;
    const int arow  = (mgroup * 16 + lrow) * FDIM;            // A row base (halves)
    const int hbase = (mgroup * 16 + koffA) * FDIM + fcol;    // h D-fragment store base

    for (int t = 0; t < DEG; ++t) {
        const int nxt = cur ^ 1;

        // issue gather of m_{t+1} into the other buffer (overlaps WMMA below)
        if (t + 1 < DEG) {
            const int e = (node0 + gnode) * DEG + (t + 1);
            const int nb = nbr[e];
            const float w = ew[e];
            const float4* src = (const float4*)(hin + (size_t)nb * rowstride + (size_t)s * FDIM + gfeat);
            float4 q0 = src[0], q1 = src[1];
            v8h hv;
            hv[0] = (_Float16)(q0.x * w); hv[1] = (_Float16)(q0.y * w);
            hv[2] = (_Float16)(q0.z * w); hv[3] = (_Float16)(q0.w * w);
            hv[4] = (_Float16)(q1.x * w); hv[5] = (_Float16)(q1.y * w);
            hv[6] = (_Float16)(q1.z * w); hv[7] = (_Float16)(q1.w * w);
            *(v8h*)(smA[nxt] + gnode * FDIM + gfeat) = hv;
            if (t + 2 < DEG) {                      // prefetch t+2's row into caches
                const int nb2 = nbr[e + 1];
                __builtin_prefetch(hin + (size_t)nb2 * rowstride + (size_t)s * FDIM + gfeat, 0, 3);
            }
        }

        // gates (i,f,g,o fragments) for this wave's 16x16 cell block
        v8f acc[4] = {{}, {}, {}, {}};
#pragma unroll
        for (int kc = 0; kc < 2; ++kc) {
            const int ka = kc * 32;
            v8h mlo = *(const v8h*)(smA[cur] + arow + ka + koffA);
            v8h mhi = *(const v8h*)(smA[cur] + arow + ka + 16 + koffA);
            v8h hlo = *(const v8h*)(shA[cur] + arow + ka + koffA);
            v8h hhi = *(const v8h*)(shA[cur] + arow + ka + 16 + koffA);
            v16h Am = cat8(mlo, mhi);
            v16h Ah = cat8(hlo, hhi);
#pragma unroll
            for (int g = 0; g < 4; ++g) {
                acc[g] = wmma16(Am, Bih[g][kc], acc[g]);
                acc[g] = wmma16(Ah, Bhh[g][kc], acc[g]);
            }
        }

        // fused LSTM cell update entirely in registers; write h (f16) to next buffer
#pragma unroll
        for (int r = 0; r < 8; ++r) {
            float ig = sigm_(acc[0][r] + sbi);
            float fg = sigm_(acc[1][r] + sbf);
            float gg = tanh_(acc[2][r] + sbg);
            float og = sigm_(acc[3][r] + sbo);
            float c  = fg * cst[r] + ig * gg;
            cst[r] = c;
            shA[nxt][hbase + r * FDIM] = (_Float16)(og * tanh_(c));
        }
        __syncthreads();
        cur = nxt;
    }

    // ---- self features into the free message buffer ----
    const int sbuf = cur ^ 1;
    {
        const float4* src = (const float4*)(hin + (size_t)(node0 + gnode) * rowstride + (size_t)s * FDIM + gfeat);
        float4 q0 = src[0], q1 = src[1];
        v8h hv;
        hv[0] = (_Float16)q0.x; hv[1] = (_Float16)q0.y;
        hv[2] = (_Float16)q0.z; hv[3] = (_Float16)q0.w;
        hv[4] = (_Float16)q1.x; hv[5] = (_Float16)q1.y;
        hv[6] = (_Float16)q1.z; hv[7] = (_Float16)q1.w;
        *(v8h*)(smA[sbuf] + gnode * FDIM + gfeat) = hv;
    }
    __syncthreads();

    // ---- out = self @ Wself^T + h_n @ Wneigh^T + b : one 16x16 tile per wave ----
    {
        const int n = fcol;                         // output column
        v8f acc = {};
#pragma unroll
        for (int kc = 0; kc < 2; ++kc) {
            const int ka = kc * 32;
            v8h slo = *(const v8h*)(smA[sbuf] + arow + ka + koffA);
            v8h shi = *(const v8h*)(smA[sbuf] + arow + ka + 16 + koffA);
            v8h nlo = *(const v8h*)(shA[cur] + arow + ka + koffA);
            v8h nhi = *(const v8h*)(shA[cur] + arow + ka + 16 + koffA);
            v16h As = cat8(slo, shi);
            v16h An = cat8(nlo, nhi);
            v16h Bs = load_row_f16(Wself  + n * FDIM + ka + koffB);
            v16h Bn = load_row_f16(Wneigh + n * FDIM + ka + koffB);
            acc = wmma16(As, Bs, acc);
            acc = wmma16(An, Bn, acc);
        }
        const float bn = bout[n];
#pragma unroll
        for (int r = 0; r < 8; ++r) {
            hout[(size_t)(node0 + mgroup * 16 + koffA + r) * rowstride + (size_t)s * FDIM + n] = acc[r] + bn;
        }
    }
}

extern "C" void kernel_launch(void* const* d_in, const int* in_sizes, int n_in,
                              void* d_out, int out_size, void* d_ws, size_t ws_size,
                              hipStream_t stream) {
    (void)in_sizes; (void)n_in; (void)out_size; (void)ws_size;

    const float* x    = (const float*)d_in[0];
    const int*   nbr1 = (const int*)  d_in[1];
    const int*   nbr2 = (const int*)  d_in[2];
    const int*   nbr3 = (const int*)  d_in[3];
    const float* ew1  = (const float*)d_in[4];
    const float* ew2  = (const float*)d_in[5];
    const float* ew3  = (const float*)d_in[6];

    // params flattened in dict order: per layer Wih, Whh, bih, bhh, Wself, Wneigh, b
    const float* P[21];
    for (int i = 0; i < 21; ++i) P[i] = (const float*)d_in[7 + i];

    float* h1 = (float*)d_ws;
    float* h2 = h1 + (size_t)NNODES * NSLICE * FDIM;
    float* out = (float*)d_out;

    dim3 grid((NNODES / 32) * NSLICE);
    dim3 block(256);

    sage_lstm_layer<<<grid, block, 0, stream>>>(x,  nbr1, ew1,
        P[0], P[1], P[2], P[3], P[4], P[5], P[6], h1);
    sage_lstm_layer<<<grid, block, 0, stream>>>(h1, nbr2, ew2,
        P[7], P[8], P[9], P[10], P[11], P[12], P[13], h2);
    sage_lstm_layer<<<grid, block, 0, stream>>>(h2, nbr3, ew3,
        P[14], P[15], P[16], P[17], P[18], P[19], P[20], out);
}
